// TermSentParaBERT_V2_Fast_42185168782012
// MI455X (gfx1250) — matrix operand order repaired and checked by reading, not compiled, not run on hardware
//
#include <hip/hip_runtime.h>
#include <hip/hip_bf16.h>
#include <math.h>

// ---- problem dims (fixed by reference) ----
#define Bn   8
#define Sn   512
#define Hn   768
#define Ln   12
#define NHn  12
#define DHn  64
#define FFn  3072
#define Pn   192
#define NTOK (Bn * Sn)   // 4096

typedef __attribute__((ext_vector_type(16))) __bf16 v16bf;
typedef __attribute__((ext_vector_type(8)))  float  v8f;

union FragU { v16bf v; uint4 q[2]; };

__device__ __forceinline__ __bf16 f2bf(float f) {
    unsigned u = __builtin_bit_cast(unsigned, f);
    unsigned r = u + 0x7FFFu + ((u >> 16) & 1u);
    unsigned short h = (unsigned short)(r >> 16);
    return __builtin_bit_cast(__bf16, h);
}
// HW packed convert: a -> [15:0], b -> [31:16]  (v_cvt_pk_bf16_f32)
__device__ __forceinline__ unsigned cvt_pk_bf16(float a, float b) {
    unsigned r;
    asm volatile("v_cvt_pk_bf16_f32 %0, %1, %2" : "=v"(r) : "v"(a), "v"(b));
    return r;
}
__device__ __forceinline__ float bf_lo(unsigned u) {
    return __builtin_bit_cast(float, u << 16);
}
__device__ __forceinline__ float bf_hi(unsigned u) {
    return __builtin_bit_cast(float, u & 0xFFFF0000u);
}

// CDNA5 async global->LDS (16B per lane), tracked by ASYNCcnt
__device__ __forceinline__ void async_g2l_b128(void* lds, const void* g) {
    asm volatile("global_load_async_to_lds_b128 %0, %1, off"
                 :: "v"((unsigned)(unsigned long long)(uintptr_t)lds),
                    "v"((unsigned long long)(uintptr_t)g)
                 : "memory");
}
__device__ __forceinline__ void wait_async0() {
    asm volatile("s_wait_asynccnt 0" ::: "memory");
}

// =====================================================================
// GEMM: C[M,N] = A_bf16[M,K] @ W_f32[K,N] + bias[N] (+res)
// 128x128x32 tiles, 8 waves (4x2), wave tile 32x64 = 8 WMMAs / K-step.
// A staged async global->LDS; W converted via v_cvt_pk_bf16_f32 into a
// transposed LDS tile so every fragment is 2x ds_load_b128.
// =====================================================================
__global__ __launch_bounds__(256) void gemm_bf16_wmma(
    const __bf16* __restrict__ A, const float* __restrict__ W,
    const float* __restrict__ bias, const float* __restrict__ residual,
    float* __restrict__ outF, __bf16* __restrict__ outB,
    int M, int N, int K)
{
    __shared__ __bf16 As[128][40];   // [row][k]  80B rows (16B mult)
    __shared__ __bf16 Bt[128][40];   // [col][k]  transposed weight tile

    const int tid  = threadIdx.x;
    const int wave = tid >> 5, lane = tid & 31;
    const int wm = wave >> 1, wn = wave & 1;       // 4x2 wave grid
    const int half = lane >> 4, ml = lane & 15;
    const int rowBase = blockIdx.y * 128;
    const int colBase = blockIdx.x * 128;

    v8f acc[2][4] = {};

    #pragma unroll 1
    for (int k0 = 0; k0 < K; k0 += 32) {
        // ---- A tile: async global->LDS, 2x16B per thread ----
        {
            int r = (tid >> 2);
            int c = (tid & 3) * 8;
            async_g2l_b128(&As[r][c],
                           A + (size_t)(rowBase + r) * K + (k0 + c));
            async_g2l_b128(&As[r + 64][c],
                           A + (size_t)(rowBase + r + 64) * K + (k0 + c));
        }
        // ---- W tile transposed: 4 coalesced loads -> 2 cvt_pk -> b64 ----
        #pragma unroll
        for (int i = 0; i < 4; i++) {
            int t  = tid + i * 256;        // 0..1023
            int n  = t & 127;
            int kq = t >> 7;               // 0..7
            const float* wp = W + (size_t)(k0 + kq * 4) * N + (colBase + n);
            float f0 = wp[0];
            float f1 = wp[(size_t)N];
            float f2 = wp[(size_t)2 * N];
            float f3 = wp[(size_t)3 * N];
            uint2 u; u.x = cvt_pk_bf16(f0, f1); u.y = cvt_pk_bf16(f2, f3);
            *reinterpret_cast<uint2*>(&Bt[n][kq * 4]) = u;
        }
        if (k0 + 32 < K) {
            __builtin_prefetch(A + (size_t)(rowBase + (tid >> 2)) * K + (k0 + 32), 0, 1);
            __builtin_prefetch(W + (size_t)(k0 + 32 + (tid >> 3)) * N + colBase + (tid & 7) * 16, 0, 1);
        }
        wait_async0();
        __syncthreads();

        v16bf af[2], bfr[4];
        #pragma unroll
        for (int mt = 0; mt < 2; mt++) {
            int r = wm * 32 + mt * 16 + ml;
            FragU u;
            u.q[0] = *reinterpret_cast<const uint4*>(&As[r][half * 8]);
            u.q[1] = *reinterpret_cast<const uint4*>(&As[r][16 + half * 8]);
            af[mt] = u.v;
        }
        #pragma unroll
        for (int nt = 0; nt < 4; nt++) {
            int c = wn * 64 + nt * 16 + ml;
            FragU u;
            u.q[0] = *reinterpret_cast<const uint4*>(&Bt[c][half * 8]);
            u.q[1] = *reinterpret_cast<const uint4*>(&Bt[c][16 + half * 8]);
            bfr[nt] = u.v;
        }
        #pragma unroll
        for (int mt = 0; mt < 2; mt++)
            #pragma unroll
            for (int nt = 0; nt < 4; nt++)
                acc[mt][nt] = __builtin_amdgcn_wmma_f32_16x16x32_bf16(
                    false, af[mt], false, bfr[nt], (short)0, acc[mt][nt],
                    false, false);
        __syncthreads();
    }

    // ---- epilogue: hoisted row pointers, immediate-offset stores ----
    const float* bp = bias + colBase + wn * 64 + ml;
    #pragma unroll
    for (int mt = 0; mt < 2; mt++) {
        #pragma unroll
        for (int r = 0; r < 8; r++) {
            int row = rowBase + wm * 32 + mt * 16 + (half ? r + 8 : r);
            size_t rowoff = (size_t)row * N + colBase + wn * 64 + ml;
            #pragma unroll
            for (int nt = 0; nt < 4; nt++) {
                float v = acc[mt][nt][r] + bp[nt * 16];
                if (residual) v += residual[rowoff + nt * 16];
                if (outF) outF[rowoff + nt * 16] = v;
                if (outB) outB[rowoff + nt * 16] = f2bf(v);
            }
        }
    }
}

// elementwise exact GELU over a bf16 buffer (in place), 4 elems/iter
__global__ __launch_bounds__(256) void gelu_bf16_kernel(
    __bf16* __restrict__ h, unsigned n4)
{
    unsigned i = blockIdx.x * 256 + threadIdx.x;
    unsigned stride = gridDim.x * 256;
    #pragma unroll 1
    for (; i < n4; i += stride) {
        uint2 u = *reinterpret_cast<uint2*>(h + (size_t)i * 4);
        float f0 = bf_lo(u.x), f1 = bf_hi(u.x);
        float f2 = bf_lo(u.y), f3 = bf_hi(u.y);
        f0 = 0.5f * f0 * (1.0f + erff(f0 * 0.70710678118f));
        f1 = 0.5f * f1 * (1.0f + erff(f1 * 0.70710678118f));
        f2 = 0.5f * f2 * (1.0f + erff(f2 * 0.70710678118f));
        f3 = 0.5f * f3 * (1.0f + erff(f3 * 0.70710678118f));
        u.x = cvt_pk_bf16(f0, f1);
        u.y = cvt_pk_bf16(f2, f3);
        *reinterpret_cast<uint2*>(h + (size_t)i * 4) = u;
    }
}

// =====================================================================
// Flash-style attention: block = (b, head, 64 q-rows), 4 waves.
// =====================================================================
__global__ __launch_bounds__(128) void attn_wmma(
    const __bf16* __restrict__ Qb, const __bf16* __restrict__ Kb,
    const __bf16* __restrict__ Vb, const int* __restrict__ amask,
    const float* __restrict__ head_bias, int layer,
    __bf16* __restrict__ ctx)
{
    const int qb = blockIdx.x & 7;            // S/64 = 8
    const int h  = (blockIdx.x >> 3) % NHn;
    const int b  = blockIdx.x / (8 * NHn);
    const int tid  = threadIdx.x;
    const int wave = tid >> 5, lane = tid & 31;
    const int half = lane >> 4, ml = lane & 15;
    const float hb    = 0.5f * head_bias[layer * NHn + h];
    const float scale = 0.125f;               // 1/sqrt(64)

    __shared__ __bf16 Ks[32][72];             // [krow][dh] 144B rows
    __shared__ __bf16 Vt[64][40];             // [dh][krow] 80B rows
    __shared__ __bf16 Ps[4][2][16][24];       // per-wave, 2 tiles, 48B rows
    __shared__ float  maskadd[32];

    v16bf aq[2];
    {
        int qrow = qb * 64 + wave * 16 + ml;
        const __bf16* qp = Qb + ((size_t)(b * Sn + qrow)) * Hn + h * DHn;
        #pragma unroll
        for (int part = 0; part < 2; part++) {
            FragU u;
            u.q[0] = *reinterpret_cast<const uint4*>(qp + part * 32 + half * 8);
            u.q[1] = *reinterpret_cast<const uint4*>(qp + part * 32 + 16 + half * 8);
            aq[part] = u.v;
        }
    }

    v8f acc[4] = {};
    float rmax[8], rsum[8];
    #pragma unroll
    for (int r = 0; r < 8; r++) { rmax[r] = -1e30f; rsum[r] = 0.0f; }

    #pragma unroll 1
    for (int kb = 0; kb < Sn / 32; kb++) {
        const int k0 = kb * 32;
        const int r = tid >> 2;
        const int c = (tid & 3) * 16;
        size_t g = ((size_t)(b * Sn + k0 + r)) * Hn + h * DHn + c;
        // K block: async global->LDS (row major)
        async_g2l_b128(&Ks[r][c],     Kb + g);
        async_g2l_b128(&Ks[r][c + 8], Kb + g + 8);
        // V block: vector load, scatter transposed into Vt
        {
            uint4 v0 = *reinterpret_cast<const uint4*>(Vb + g);
            uint4 v1 = *reinterpret_cast<const uint4*>(Vb + g + 8);
            const __bf16* e0 = reinterpret_cast<const __bf16*>(&v0);
            const __bf16* e1 = reinterpret_cast<const __bf16*>(&v1);
            #pragma unroll
            for (int j = 0; j < 8; j++) {
                Vt[c + j][r]     = e0[j];
                Vt[c + 8 + j][r] = e1[j];
            }
        }
        if (tid < 32)
            maskadd[tid] = (1.0f - (float)amask[b * Sn + k0 + tid]) * -10000.0f;
        wait_async0();
        __syncthreads();

        // scores: two 16x16 tiles, K-dim = 64 (2 chained WMMAs each)
        v8f sc[2] = {};
        #pragma unroll
        for (int j = 0; j < 2; j++) {
            #pragma unroll
            for (int part = 0; part < 2; part++) {
                const __bf16* kp = &Ks[j * 16 + ml][part * 32 + half * 8];
                FragU u;
                u.q[0] = *reinterpret_cast<const uint4*>(kp);
                u.q[1] = *reinterpret_cast<const uint4*>(kp + 16);
                sc[j] = __builtin_amdgcn_wmma_f32_16x16x32_bf16(
                    false, aq[part], false, u.v, (short)0, sc[j], false, false);
            }
        }

        // online softmax (row stats per 16-lane half)
        float pv0[8], pv1[8];
        #pragma unroll
        for (int r8 = 0; r8 < 8; r8++) {
            float s0 = sc[0][r8] * scale + maskadd[ml]      + hb;
            float s1 = sc[1][r8] * scale + maskadd[16 + ml] + hb;
            float t = fmaxf(s0, s1);
            t = fmaxf(t, __shfl_xor(t, 1, 32));
            t = fmaxf(t, __shfl_xor(t, 2, 32));
            t = fmaxf(t, __shfl_xor(t, 4, 32));
            t = fmaxf(t, __shfl_xor(t, 8, 32));
            float nm   = fmaxf(rmax[r8], t);
            float corr = __expf(rmax[r8] - nm);
            float p0 = __expf(s0 - nm), p1 = __expf(s1 - nm);
            float ps = p0 + p1;
            ps += __shfl_xor(ps, 1, 32);
            ps += __shfl_xor(ps, 2, 32);
            ps += __shfl_xor(ps, 4, 32);
            ps += __shfl_xor(ps, 8, 32);
            rsum[r8] = rsum[r8] * corr + ps;
            rmax[r8] = nm;
            #pragma unroll
            for (int nt = 0; nt < 4; nt++) acc[nt][r8] *= corr;
            pv0[r8] = p0; pv1[r8] = p1;
        }

        // P tiles -> per-wave LDS (transpose C-layout -> A-layout)
        #pragma unroll
        for (int r8 = 0; r8 < 8; r8++) {
            int row = half * 8 + r8;
            Ps[wave][0][row][ml] = f2bf(pv0[r8]);
            Ps[wave][1][row][ml] = f2bf(pv1[r8]);
        }
        v16bf ap;
        {
            FragU u;   // same-wave LDS ops are in-order
            u.q[0] = *reinterpret_cast<const uint4*>(&Ps[wave][0][ml][half * 8]);
            u.q[1] = *reinterpret_cast<const uint4*>(&Ps[wave][1][ml][half * 8]);
            ap = u.v;
        }

        // context: acc += P(16x32) @ V(32x64), 4 WMMAs
        #pragma unroll
        for (int nt = 0; nt < 4; nt++) {
            const __bf16* vp = &Vt[nt * 16 + ml][half * 8];
            FragU u;
            u.q[0] = *reinterpret_cast<const uint4*>(vp);
            u.q[1] = *reinterpret_cast<const uint4*>(vp + 16);
            acc[nt] = __builtin_amdgcn_wmma_f32_16x16x32_bf16(
                false, ap, false, u.v, (short)0, acc[nt], false, false);
        }
        __syncthreads();
    }

    // normalize + write context (bf16)
    #pragma unroll
    for (int r8 = 0; r8 < 8; r8++) {
        float inv = 1.0f / rsum[r8];
        int qrow = qb * 64 + wave * 16 + (half ? r8 + 8 : r8);
        size_t base = ((size_t)(b * Sn + qrow)) * Hn + h * DHn;
        #pragma unroll
        for (int nt = 0; nt < 4; nt++)
            ctx[base + nt * 16 + ml] = f2bf(acc[nt][r8] * inv);
    }
}

// =====================================================================
// Embedding gather + LayerNorm (writes f32 + bf16 activations)
// =====================================================================
__global__ __launch_bounds__(256) void embed_ln(
    const int* __restrict__ ids, const int* __restrict__ tts,
    const float* __restrict__ we, const float* __restrict__ pe,
    const float* __restrict__ te, const float* __restrict__ g,
    const float* __restrict__ beta, float* __restrict__ outF,
    __bf16* __restrict__ outB)
{
    __shared__ float red[256];
    const int n = blockIdx.x, s = n % Sn;
    const int id = ids[n], tt = tts[n];
    float v[3], sum = 0.0f;
    #pragma unroll
    for (int i = 0; i < 3; i++) {
        int e = threadIdx.x + i * 256;
        v[i] = we[(size_t)id * Hn + e] + pe[(size_t)s * Hn + e] +
               te[(size_t)tt * Hn + e];
        sum += v[i];
    }
    red[threadIdx.x] = sum; __syncthreads();
    for (int o = 128; o > 0; o >>= 1) {
        if (threadIdx.x < o) red[threadIdx.x] += red[threadIdx.x + o];
        __syncthreads();
    }
    float mean = red[0] * (1.0f / Hn); __syncthreads();
    float sq = 0.0f;
    #pragma unroll
    for (int i = 0; i < 3; i++) { float d = v[i] - mean; sq += d * d; }
    red[threadIdx.x] = sq; __syncthreads();
    for (int o = 128; o > 0; o >>= 1) {
        if (threadIdx.x < o) red[threadIdx.x] += red[threadIdx.x + o];
        __syncthreads();
    }
    float rstd = rsqrtf(red[0] * (1.0f / Hn) + 1e-12f);
    #pragma unroll
    for (int i = 0; i < 3; i++) {
        int e = threadIdx.x + i * 256;
        float y = g[e] * ((v[i] - mean) * rstd) + beta[e];
        outF[(size_t)n * Hn + e] = y;
        outB[(size_t)n * Hn + e] = f2bf(y);
    }
}

__global__ __launch_bounds__(256) void ln_kernel(
    const float* __restrict__ in, const float* __restrict__ g,
    const float* __restrict__ beta, float* __restrict__ outF,
    __bf16* __restrict__ outB)
{
    __shared__ float red[256];
    const int n = blockIdx.x;
    float v[3], sum = 0.0f;
    #pragma unroll
    for (int i = 0; i < 3; i++) {
        v[i] = in[(size_t)n * Hn + threadIdx.x + i * 256];
        sum += v[i];
    }
    red[threadIdx.x] = sum; __syncthreads();
    for (int o = 128; o > 0; o >>= 1) {
        if (threadIdx.x < o) red[threadIdx.x] += red[threadIdx.x + o];
        __syncthreads();
    }
    float mean = red[0] * (1.0f / Hn); __syncthreads();
    float sq = 0.0f;
    #pragma unroll
    for (int i = 0; i < 3; i++) { float d = v[i] - mean; sq += d * d; }
    red[threadIdx.x] = sq; __syncthreads();
    for (int o = 128; o > 0; o >>= 1) {
        if (threadIdx.x < o) red[threadIdx.x] += red[threadIdx.x + o];
        __syncthreads();
    }
    float rstd = rsqrtf(red[0] * (1.0f / Hn) + 1e-12f);
    #pragma unroll
    for (int i = 0; i < 3; i++) {
        int e = threadIdx.x + i * 256;
        float y = g[e] * ((v[i] - mean) * rstd) + beta[e];
        outF[(size_t)n * Hn + e] = y;
        outB[(size_t)n * Hn + e] = f2bf(y);
    }
}

// ---- small tail kernels ----
__global__ __launch_bounds__(256) void pooler_kernel(
    const float* __restrict__ seq, const float* __restrict__ pw,
    const float* __restrict__ pb, float* __restrict__ pooled)
{
    int b = blockIdx.x;
    #pragma unroll 1
    for (int i = 0; i < 3; i++) {
        int col = threadIdx.x + i * 256;
        float a = pb[col];
        #pragma unroll 4
        for (int k = 0; k < Hn; k++)
            a += seq[(size_t)b * Sn * Hn + k] * pw[(size_t)k * Hn + col];
        pooled[b * Hn + col] = tanhf(a);
    }
}

__global__ __launch_bounds__(256) void term_bias_kernel(
    const float* __restrict__ seq, const float* __restrict__ tbw,
    const float* __restrict__ tbb, const int* __restrict__ term_mask,
    const int* __restrict__ amask, float* __restrict__ stb)
{
    __shared__ float red[256];
    int n = blockIdx.x;
    float s = 0.0f;
    #pragma unroll
    for (int i = 0; i < 3; i++) {
        int e = threadIdx.x + i * 256;
        s += seq[(size_t)n * Hn + e] * tbw[e];
    }
    red[threadIdx.x] = s; __syncthreads();
    for (int o = 128; o > 0; o >>= 1) {
        if (threadIdx.x < o) red[threadIdx.x] += red[threadIdx.x + o];
        __syncthreads();
    }
    if (threadIdx.x == 0) {
        float rt = red[0] + tbb[0];
        float tb = term_mask[n] ? (tanhf(rt) * 2.0f) : -5.0f;
        stb[n] = tb + (1.0f - (float)amask[n]) * -10000.0f;
    }
}

__global__ __launch_bounds__(256) void term_softmax_kernel(
    const float* __restrict__ stb, float* __restrict__ wsm)
{
    __shared__ float red[256];
    int b = blockIdx.x;
    float x0 = stb[b * Sn + threadIdx.x];
    float x1 = stb[b * Sn + threadIdx.x + 256];
    red[threadIdx.x] = fmaxf(x0, x1); __syncthreads();
    for (int o = 128; o > 0; o >>= 1) {
        if (threadIdx.x < o)
            red[threadIdx.x] = fmaxf(red[threadIdx.x], red[threadIdx.x + o]);
        __syncthreads();
    }
    float mx = red[0]; __syncthreads();
    float e0 = __expf(x0 - mx), e1 = __expf(x1 - mx);
    red[threadIdx.x] = e0 + e1; __syncthreads();
    for (int o = 128; o > 0; o >>= 1) {
        if (threadIdx.x < o) red[threadIdx.x] += red[threadIdx.x + o];
        __syncthreads();
    }
    float inv = 1.0f / red[0];
    wsm[b * Sn + threadIdx.x]       = e0 * inv;
    wsm[b * Sn + threadIdx.x + 256] = e1 * inv;
}

__global__ __launch_bounds__(256) void term_enh_kernel(
    const float* __restrict__ seq, const float* __restrict__ wsm,
    float* __restrict__ enh)
{
    int b = blockIdx.x / 3;
    int h = (blockIdx.x % 3) * 256 + threadIdx.x;
    float a = 0.0f;
    #pragma unroll 4
    for (int s = 0; s < Sn; s++)
        a += seq[((size_t)b * Sn + s) * Hn + h] * wsm[b * Sn + s];
    enh[b * Hn + h] = a;
}

__global__ __launch_bounds__(192) void term_proj_kernel(
    const float* __restrict__ enh, const float* __restrict__ tpw,
    const float* __restrict__ tpb, float* __restrict__ tf)
{
    int b = blockIdx.x, col = threadIdx.x;
    float a = tpb[col];
    #pragma unroll 4
    for (int k = 0; k < Hn; k++)
        a += enh[b * Hn + k] * tpw[(size_t)k * Pn + col];
    tf[b * Pn + col] = a;
}

__global__ __launch_bounds__(16) void cls_kernel(
    const float* __restrict__ pooled, const float* __restrict__ tfeat,
    const float* __restrict__ cls_w, const float* __restrict__ cls_b,
    float* __restrict__ out)
{
    int b = threadIdx.x >> 1, c = threadIdx.x & 1;
    float a = cls_b[c];
    for (int i = 0; i < Hn; i++) a += pooled[b * Hn + i] * cls_w[i * 2 + c];
    for (int i = 0; i < Pn; i++) a += tfeat[b * Pn + i] * cls_w[(Hn + i) * 2 + c];
    out[b * 2 + c] = a;
}

// =====================================================================
// host orchestration
// =====================================================================
extern "C" void kernel_launch(void* const* d_in, const int* in_sizes, int n_in,
                              void* d_out, int out_size, void* d_ws, size_t ws_size,
                              hipStream_t stream) {
    (void)in_sizes; (void)n_in; (void)out_size; (void)ws_size;
    const int*   input_ids  = (const int*)d_in[0];
    const int*   amask      = (const int*)d_in[1];
    const int*   ttids      = (const int*)d_in[2];
    const int*   term_mask  = (const int*)d_in[3];
    const float* word_emb   = (const float*)d_in[4];
    const float* pos_emb    = (const float*)d_in[5];
    const float* type_emb   = (const float*)d_in[6];
    const float* emb_ln_g   = (const float*)d_in[7];
    const float* emb_ln_b   = (const float*)d_in[8];
    const float* wq         = (const float*)d_in[9];
    const float* bq         = (const float*)d_in[10];
    const float* wk         = (const float*)d_in[11];
    const float* bk         = (const float*)d_in[12];
    const float* wv         = (const float*)d_in[13];
    const float* bv         = (const float*)d_in[14];
    const float* head_bias  = (const float*)d_in[15];
    const float* aow        = (const float*)d_in[16];
    const float* aob        = (const float*)d_in[17];
    const float* ln1g       = (const float*)d_in[18];
    const float* ln1b       = (const float*)d_in[19];
    const float* fw1        = (const float*)d_in[20];
    const float* fb1        = (const float*)d_in[21];
    const float* fw2        = (const float*)d_in[22];
    const float* fb2        = (const float*)d_in[23];
    const float* ln2g       = (const float*)d_in[24];
    const float* ln2b       = (const float*)d_in[25];
    const float* pooler_w   = (const float*)d_in[26];
    const float* pooler_b   = (const float*)d_in[27];
    const float* tbw        = (const float*)d_in[28];
    const float* tbb        = (const float*)d_in[29];
    const float* tpw        = (const float*)d_in[30];
    const float* tpb        = (const float*)d_in[31];
    const float* cls_w      = (const float*)d_in[32];
    const float* cls_b      = (const float*)d_in[33];

    char* wptr = (char*)d_ws;
    auto carve = [&](size_t bytes) -> void* {
        void* p = (void*)wptr;
        wptr += (bytes + 255) & ~(size_t)255;
        return p;
    };
    float*  xF    = (float*)carve((size_t)NTOK * Hn * 4);
    float*  tmpF  = (float*)carve((size_t)NTOK * Hn * 4);
    __bf16* xB    = (__bf16*)carve((size_t)NTOK * Hn * 2);
    __bf16* qB    = (__bf16*)carve((size_t)NTOK * Hn * 2);
    __bf16* kB    = (__bf16*)carve((size_t)NTOK * Hn * 2);
    __bf16* vB    = (__bf16*)carve((size_t)NTOK * Hn * 2);
    __bf16* ctxB  = (__bf16*)carve((size_t)NTOK * Hn * 2);
    __bf16* hB    = (__bf16*)carve((size_t)NTOK * FFn * 2);
    float*  stb   = (float*)carve((size_t)NTOK * 4);
    float*  wsm   = (float*)carve((size_t)NTOK * 4);
    float*  pooled= (float*)carve((size_t)Bn * Hn * 4);
    float*  enh   = (float*)carve((size_t)Bn * Hn * 4);
    float*  tfeat = (float*)carve((size_t)Bn * Pn * 4);

    embed_ln<<<NTOK, 256, 0, stream>>>(input_ids, ttids, word_emb, pos_emb,
                                       type_emb, emb_ln_g, emb_ln_b, xF, xB);

    const dim3 gH(Hn / 128, NTOK / 128);    // (6, 32)
    const dim3 gFF(FFn / 128, NTOK / 128);  // (24, 32)

    for (int i = 0; i < Ln; i++) {
        const float* wq_i = wq + (size_t)i * Hn * Hn;
        const float* wk_i = wk + (size_t)i * Hn * Hn;
        const float* wv_i = wv + (size_t)i * Hn * Hn;
        const float* ao_i = aow + (size_t)i * Hn * Hn;
        const float* w1_i = fw1 + (size_t)i * Hn * FFn;
        const float* w2_i = fw2 + (size_t)i * FFn * Hn;

        gemm_bf16_wmma<<<gH, 256, 0, stream>>>(xB, wq_i, bq + i * Hn, nullptr,
                                               nullptr, qB, NTOK, Hn, Hn);
        gemm_bf16_wmma<<<gH, 256, 0, stream>>>(xB, wk_i, bk + i * Hn, nullptr,
                                               nullptr, kB, NTOK, Hn, Hn);
        gemm_bf16_wmma<<<gH, 256, 0, stream>>>(xB, wv_i, bv + i * Hn, nullptr,
                                               nullptr, vB, NTOK, Hn, Hn);
        attn_wmma<<<Bn * NHn * (Sn / 64), 128, 0, stream>>>(
            qB, kB, vB, amask, head_bias, i, ctxB);
        gemm_bf16_wmma<<<gH, 256, 0, stream>>>(ctxB, ao_i, aob + i * Hn, xF,
                                               tmpF, nullptr, NTOK, Hn, Hn);
        ln_kernel<<<NTOK, 256, 0, stream>>>(tmpF, ln1g + i * Hn, ln1b + i * Hn,
                                            xF, xB);
        gemm_bf16_wmma<<<gFF, 256, 0, stream>>>(xB, w1_i, fb1 + i * FFn, nullptr,
                                                nullptr, hB, NTOK, FFn, Hn);
        gelu_bf16_kernel<<<2048, 256, 0, stream>>>(
            hB, (unsigned)((size_t)NTOK * FFn / 4));
        gemm_bf16_wmma<<<gH, 256, 0, stream>>>(hB, w2_i, fb2 + i * Hn, xF,
                                               tmpF, nullptr, NTOK, Hn, FFn);
        ln_kernel<<<NTOK, 256, 0, stream>>>(tmpF, ln2g + i * Hn, ln2b + i * Hn,
                                            xF, xB);
    }

    pooler_kernel<<<Bn, 256, 0, stream>>>(xF, pooler_w, pooler_b, pooled);
    term_bias_kernel<<<NTOK, 256, 0, stream>>>(xF, tbw, tbb, term_mask, amask, stb);
    term_softmax_kernel<<<Bn, 256, 0, stream>>>(stb, wsm);
    term_enh_kernel<<<Bn * 3, 256, 0, stream>>>(xF, wsm, enh);
    term_proj_kernel<<<Bn, 192, 0, stream>>>(enh, tpw, tpb, tfeat);
    cls_kernel<<<1, 16, 0, stream>>>(pooled, tfeat, cls_w, cls_b, (float*)d_out);
}
